// mdLSTM_55104430407701
// MI455X (gfx1250) — matrix operand order repaired and checked by reading, not compiled, not run on hardware
//
#include <hip/hip_runtime.h>
#include <hip/hip_bf16.h>
#include <hip/hip_fp16.h>

typedef __attribute__((ext_vector_type(16))) _Float16 v16h;
typedef __attribute__((ext_vector_type(8)))  float    v8f;

#define S_    1024
#define T_    96
#define P_    48
#define IN_   64
#define EMBH_ 128
#define HID_  256
#define G4_   1024
#define KCAT_ 320

// ---------------------------------------------------------------------------
// WMMA fragment loaders (gfx1250 wave32, V_WMMA_F32_16X16X32_F16)
// A: 16x32 f16, row m = lane&15. Lanes 0-15 hold K {k0..k0+7, k0+16..k0+23},
//    lanes 16-31 hold K {k0+8..k0+15, k0+24..k0+31}. Two contiguous b128 loads.
// B: 32x16 f16 == W(N,K) row-major rows. Lane holds the same K split as A, of row n.
// ---------------------------------------------------------------------------
__device__ __forceinline__ v16h ld_a(const _Float16* __restrict__ A, int lda,
                                     int row0, int k0) {
  const int lane = threadIdx.x & 31;
  const _Float16* p =
      A + (size_t)(row0 + (lane & 15)) * lda + k0 + ((lane >> 4) << 3);
  v16h f;
  float4* d = reinterpret_cast<float4*>(&f);
  d[0] = *reinterpret_cast<const float4*>(p);
  d[1] = *reinterpret_cast<const float4*>(p + 16);
  return f;
}

__device__ __forceinline__ v16h ld_b(const _Float16* __restrict__ W, int ldk,
                                     int n0, int k0) {
  const int lane = threadIdx.x & 31;
  const _Float16* p =
      W + (size_t)(n0 + (lane & 15)) * ldk + k0 + ((lane >> 4) << 3);
  v16h f;
  float4* d = reinterpret_cast<float4*>(&f);
  d[0] = *reinterpret_cast<const float4*>(p);
  d[1] = *reinterpret_cast<const float4*>(p + 16);
  return f;
}

__device__ __forceinline__ v8f wmma_acc(v16h a, v16h b, v8f c) {
  return __builtin_amdgcn_wmma_f32_16x16x32_f16(false, a, false, b, (short)0, c,
                                                false, false);
}

__device__ __forceinline__ float sigf(float x) { return 1.f / (1.f + __expf(-x)); }

// ---------------------------------------------------------------------------
// Generic register-blocked GEMM: 32x64 output tile per wave (2M x 4N subtiles)
//   O(M,N) f16 = [relu]( A(M,K) f16 * B(N,K)^T f16 + bias )
// Requires M%32==0, N%64==0, K%32==0.
// Per K-step: 2 A-frags + 4 B-frags (12 b128 loads) feed 8 independent WMMAs.
// ---------------------------------------------------------------------------
__global__ __launch_bounds__(256) void k_gemm(
    const _Float16* __restrict__ A, int lda,
    const _Float16* __restrict__ B, int ldb,
    const float* __restrict__ bias,
    _Float16* __restrict__ O, int ldo,
    int M, int N, int K, int relu) {
  const int wave = (int)((blockIdx.x * blockDim.x + threadIdx.x) >> 5);
  const int ng = N >> 6;
  if (wave >= (M >> 5) * ng) return;
  const int tm = (wave / ng) << 5;
  const int tn = (wave % ng) << 6;
  v8f acc[2][4] = {};
  for (int k = 0; k < K; k += 32) {
    v16h a0 = ld_a(A, lda, tm, k);
    v16h a1 = ld_a(A, lda, tm + 16, k);
    v16h b0 = ld_b(B, ldb, tn, k);
    v16h b1 = ld_b(B, ldb, tn + 16, k);
    v16h b2 = ld_b(B, ldb, tn + 32, k);
    v16h b3 = ld_b(B, ldb, tn + 48, k);
    acc[0][0] = wmma_acc(a0, b0, acc[0][0]);
    acc[0][1] = wmma_acc(a0, b1, acc[0][1]);
    acc[0][2] = wmma_acc(a0, b2, acc[0][2]);
    acc[0][3] = wmma_acc(a0, b3, acc[0][3]);
    acc[1][0] = wmma_acc(a1, b0, acc[1][0]);
    acc[1][1] = wmma_acc(a1, b1, acc[1][1]);
    acc[1][2] = wmma_acc(a1, b2, acc[1][2]);
    acc[1][3] = wmma_acc(a1, b3, acc[1][3]);
  }
  const int lane = threadIdx.x & 31;
#pragma unroll
  for (int mi = 0; mi < 2; ++mi) {
#pragma unroll
    for (int ni = 0; ni < 4; ++ni) {
      const int n = tn + (ni << 4) + (lane & 15);
      const int m0 = tm + (mi << 4) + ((lane >> 4) << 3);
      const float bv = bias ? bias[n] : 0.f;
#pragma unroll
      for (int r = 0; r < 8; ++r) {
        float v = acc[mi][ni][r] + bv;
        if (relu) v = v > 0.f ? v : 0.f;
        O[(size_t)(m0 + r) * ldo + n] = (_Float16)v;
      }
    }
  }
}

// ---------------------------------------------------------------------------
// Step GEMM: gates(S,1024) f32 = [x_t | h] (S,320) * Wcat(1024,320)^T + gbias
// 32x64 register-blocked tile per wave. x_t rows at stride T_*IN_, h at HID_.
// ---------------------------------------------------------------------------
__global__ __launch_bounds__(256) void k_gates(
    const _Float16* __restrict__ Xt,   // xemb + t*IN_, lda = T_*IN_
    const _Float16* __restrict__ H,    // (S, HID_) f16, post-conv h of prev step
    const _Float16* __restrict__ Wc,   // (1024, 320) f16
    const float* __restrict__ gb,      // b_ih + b_hh
    float* __restrict__ G) {           // (S, 1024) f32
  const int wave = (int)((blockIdx.x * blockDim.x + threadIdx.x) >> 5);
  if (wave >= (S_ >> 5) * (G4_ >> 6)) return;
  const int tm = (wave >> 4) << 5;    // 32 row-groups
  const int tn = (wave & 15) << 6;    // 16 col-groups of 64
  v8f acc[2][4] = {};
#pragma unroll
  for (int k = 0; k < KCAT_; k += 32) {
    const _Float16* Ak = (k < IN_) ? Xt : H;
    const int lda = (k < IN_) ? T_ * IN_ : HID_;
    const int kk = (k < IN_) ? k : k - IN_;
    v16h a0 = ld_a(Ak, lda, tm, kk);
    v16h a1 = ld_a(Ak, lda, tm + 16, kk);
    v16h b0 = ld_b(Wc, KCAT_, tn, k);
    v16h b1 = ld_b(Wc, KCAT_, tn + 16, k);
    v16h b2 = ld_b(Wc, KCAT_, tn + 32, k);
    v16h b3 = ld_b(Wc, KCAT_, tn + 48, k);
    acc[0][0] = wmma_acc(a0, b0, acc[0][0]);
    acc[0][1] = wmma_acc(a0, b1, acc[0][1]);
    acc[0][2] = wmma_acc(a0, b2, acc[0][2]);
    acc[0][3] = wmma_acc(a0, b3, acc[0][3]);
    acc[1][0] = wmma_acc(a1, b0, acc[1][0]);
    acc[1][1] = wmma_acc(a1, b1, acc[1][1]);
    acc[1][2] = wmma_acc(a1, b2, acc[1][2]);
    acc[1][3] = wmma_acc(a1, b3, acc[1][3]);
  }
  const int lane = threadIdx.x & 31;
#pragma unroll
  for (int mi = 0; mi < 2; ++mi) {
#pragma unroll
    for (int ni = 0; ni < 4; ++ni) {
      const int n = tn + (ni << 4) + (lane & 15);
      const int m0 = tm + (mi << 4) + ((lane >> 4) << 3);
      const float bv = gb[n];
#pragma unroll
      for (int r = 0; r < 8; ++r)
        G[(size_t)(m0 + r) * G4_ + n] = acc[mi][ni][r] + bv;
    }
  }
}

// ---------------------------------------------------------------------------
// LSTM cell pointwise (torch gate order i,f,g,o).
// ---------------------------------------------------------------------------
__global__ __launch_bounds__(256) void k_cell(
    const float* __restrict__ G, float* __restrict__ c,
    float* __restrict__ Cout, _Float16* __restrict__ hcpad, int t) {
  const int idx = blockIdx.x * blockDim.x + threadIdx.x;
  if (idx >= S_ * HID_) return;
  const int s = idx >> 8, ch = idx & 255;
  const float* g = G + (size_t)s * G4_;
  const float ig = sigf(g[ch]);
  const float fg = sigf(g[ch + 256]);
  const float gg = tanhf(g[ch + 512]);
  const float og = sigf(g[ch + 768]);
  const float cn = fg * c[idx] + ig * gg;
  c[idx] = cn;
  Cout[(size_t)idx * T_ + t] = cn;                       // C[s,ch,t]
  hcpad[(size_t)(s + 1) * HID_ + ch] = (_Float16)(og * tanhf(cn));
}

// ---------------------------------------------------------------------------
// Spatial conv over samples as 3 accumulating GEMMs on zero-padded h_cell:
// h[s,o] = cb[o] + sum_k sum_i Wk[o,i] * hcpad[s+k, i]   (k = 0..2)
// 32x64 register-blocked tile per wave.
// ---------------------------------------------------------------------------
__global__ __launch_bounds__(256) void k_conv(
    const _Float16* __restrict__ hcpad,  // (S+2, HID) f16
    const _Float16* __restrict__ W3,     // 3 x (HID,HID) f16
    const float* __restrict__ cb,
    _Float16* __restrict__ hout,         // (S, HID) f16
    _Float16* __restrict__ Hp,           // (S*P_, HID) f16 head input
    float* __restrict__ Hout, int t) {
  const int wave = (int)((blockIdx.x * blockDim.x + threadIdx.x) >> 5);
  if (wave >= (S_ >> 5) * (HID_ >> 6)) return;
  const int tm = (wave >> 2) << 5;
  const int tn = (wave & 3) << 6;
  v8f acc[2][4] = {};
#pragma unroll
  for (int sh = 0; sh < 3; ++sh) {
    const _Float16* A = hcpad + sh * HID_;
    const _Float16* B = W3 + (size_t)sh * HID_ * HID_;
    for (int k = 0; k < HID_; k += 32) {
      v16h a0 = ld_a(A, HID_, tm, k);
      v16h a1 = ld_a(A, HID_, tm + 16, k);
      v16h b0 = ld_b(B, HID_, tn, k);
      v16h b1 = ld_b(B, HID_, tn + 16, k);
      v16h b2 = ld_b(B, HID_, tn + 32, k);
      v16h b3 = ld_b(B, HID_, tn + 48, k);
      acc[0][0] = wmma_acc(a0, b0, acc[0][0]);
      acc[0][1] = wmma_acc(a0, b1, acc[0][1]);
      acc[0][2] = wmma_acc(a0, b2, acc[0][2]);
      acc[0][3] = wmma_acc(a0, b3, acc[0][3]);
      acc[1][0] = wmma_acc(a1, b0, acc[1][0]);
      acc[1][1] = wmma_acc(a1, b1, acc[1][1]);
      acc[1][2] = wmma_acc(a1, b2, acc[1][2]);
      acc[1][3] = wmma_acc(a1, b3, acc[1][3]);
    }
  }
  const int lane = threadIdx.x & 31;
#pragma unroll
  for (int mi = 0; mi < 2; ++mi) {
#pragma unroll
    for (int ni = 0; ni < 4; ++ni) {
      const int n = tn + (ni << 4) + (lane & 15);
      const int m0 = tm + (mi << 4) + ((lane >> 4) << 3);
      const float bv = cb[n];
#pragma unroll
      for (int r = 0; r < 8; ++r) {
        const int s = m0 + r;
        const float v = acc[mi][ni][r] + bv;
        hout[(size_t)s * HID_ + n] = (_Float16)v;
        Hout[((size_t)s * HID_ + n) * T_ + t] = v;       // H[s,ch,t]
        if (t >= P_)
          Hp[((size_t)s * P_ + (t - P_)) * HID_ + n] = (_Float16)v;
      }
    }
  }
}

// ---------------------------------------------------------------------------
// Small pointwise / packing kernels
// ---------------------------------------------------------------------------
__global__ void k_lane(const float* __restrict__ lane,
                       const float* __restrict__ w1, const float* __restrict__ b1,
                       const float* __restrict__ w2, const float* __restrict__ b2,
                       float* __restrict__ lc) {
  const int s = blockIdx.x * blockDim.x + threadIdx.x;
  if (s >= S_) return;
  const float x = lane[s];
  float acc = b2[0];
#pragma unroll
  for (int j = 0; j < 32; ++j) {
    const float u = x * w1[j] + b1[j];
    acc += (u > 0.f ? u : 0.f) * w2[j];
  }
  lc[s] = sigf(acc);
}

__global__ void k_scale(const float* __restrict__ x, const float* __restrict__ lc,
                        _Float16* __restrict__ xin) {
  const int idx = blockIdx.x * blockDim.x + threadIdx.x;
  if (idx >= S_ * T_ * IN_) return;
  xin[idx] = (_Float16)(x[idx] * lc[idx / (T_ * IN_)]);
}

__global__ void k_pack16(const float* __restrict__ src, _Float16* __restrict__ dst,
                         int n) {
  const int i = blockIdx.x * blockDim.x + threadIdx.x;
  if (i < n) dst[i] = (_Float16)src[i];
}

__global__ void k_packwcat(const float* __restrict__ Wih,
                           const float* __restrict__ Whh,
                           _Float16* __restrict__ Wc) {
  const int i = blockIdx.x * blockDim.x + threadIdx.x;
  if (i >= G4_ * KCAT_) return;
  const int n = i / KCAT_, k = i % KCAT_;
  const float v = (k < IN_) ? Wih[(size_t)n * IN_ + k]
                            : Whh[(size_t)n * HID_ + (k - IN_)];
  Wc[i] = (_Float16)v;
}

__global__ void k_packconv(const float* __restrict__ cw, _Float16* __restrict__ W3) {
  const int i = blockIdx.x * blockDim.x + threadIdx.x;
  if (i >= 3 * HID_ * HID_) return;
  const int kk = i >> 16, rem = i & 65535;
  const int ch = rem >> 8, ci = rem & 255;
  W3[i] = (_Float16)cw[(size_t)ch * (HID_ * 3) + ci * 3 + kk];
}

__global__ void k_gbias(const float* __restrict__ bi, const float* __restrict__ bh,
                        float* __restrict__ gb) {
  const int i = blockIdx.x * blockDim.x + threadIdx.x;
  if (i < G4_) gb[i] = bi[i] + bh[i];
}

__global__ void k_init(_Float16* __restrict__ hbuf, float* __restrict__ c,
                       _Float16* __restrict__ hcpad) {
  const int i = blockIdx.x * blockDim.x + threadIdx.x;
  if (i >= S_ * HID_) return;
  hbuf[i] = (_Float16)0.f;
  c[i] = 0.f;
  if (i < HID_) {
    hcpad[i] = (_Float16)0.f;                               // pad row 0
    hcpad[(size_t)(S_ + 1) * HID_ + i] = (_Float16)0.f;     // pad row S+1
  }
}

__global__ void k_final(const _Float16* __restrict__ B3,
                        const float* __restrict__ w3, const float* __restrict__ b3,
                        const float* __restrict__ lc, float* __restrict__ out) {
  const int r = blockIdx.x * blockDim.x + threadIdx.x;
  if (r >= S_ * P_) return;
  const _Float16* p = B3 + (size_t)r * 64;
  float acc = b3[0];
#pragma unroll
  for (int i = 0; i < 64; ++i) acc += (float)p[i] * w3[i];
  out[r] = acc / lc[r / P_];
}

// ---------------------------------------------------------------------------
extern "C" void kernel_launch(void* const* d_in, const int* in_sizes, int n_in,
                              void* d_out, int out_size, void* d_ws, size_t ws_size,
                              hipStream_t stream) {
  const float* inputData = (const float*)d_in[0];
  const float* lane      = (const float*)d_in[1];
  const float* lg_w1 = (const float*)d_in[2];  const float* lg_b1 = (const float*)d_in[3];
  const float* lg_w2 = (const float*)d_in[4];  const float* lg_b2 = (const float*)d_in[5];
  const float* emb_w1= (const float*)d_in[6];  const float* emb_b1= (const float*)d_in[7];
  const float* emb_w2= (const float*)d_in[8];  const float* emb_b2= (const float*)d_in[9];
  const float* W_ih  = (const float*)d_in[10]; const float* W_hh  = (const float*)d_in[11];
  const float* b_ih  = (const float*)d_in[12]; const float* b_hh  = (const float*)d_in[13];
  const float* conv_w= (const float*)d_in[14]; const float* conv_b= (const float*)d_in[15];
  const float* out_w = (const float*)d_in[16]; const float* out_b = (const float*)d_in[17];
  const float* o1_w  = (const float*)d_in[18]; const float* o1_b  = (const float*)d_in[19];
  const float* o2_w  = (const float*)d_in[20]; const float* o2_b  = (const float*)d_in[21];
  const float* o3_w  = (const float*)d_in[22]; const float* o3_b  = (const float*)d_in[23];

  float* out  = (float*)d_out;                 // (S, P)
  float* Hout = out + (size_t)S_ * P_;         // (S, HID, T)
  float* Cout = Hout + (size_t)S_ * HID_ * T_; // (S, HID, T)

  char* cur = (char*)d_ws;
  auto alloc = [&](size_t bytes) -> void* {
    void* p = (void*)cur;
    cur += (bytes + 255) & ~(size_t)255;
    return p;
  };
  const size_t ROWS = (size_t)S_ * T_;   // 98304
  const size_t HR   = (size_t)S_ * P_;   // 49152
  float*    lc    = (float*)alloc(S_ * 4);
  float*    gb    = (float*)alloc(G4_ * 4);
  _Float16* Wcat  = (_Float16*)alloc((size_t)G4_ * KCAT_ * 2);
  _Float16* W3    = (_Float16*)alloc((size_t)3 * HID_ * HID_ * 2);
  _Float16* We1   = (_Float16*)alloc((size_t)EMBH_ * IN_ * 2);
  _Float16* We2   = (_Float16*)alloc((size_t)IN_ * EMBH_ * 2);
  _Float16* Wo    = (_Float16*)alloc((size_t)64 * HID_ * 2);
  _Float16* W1h   = (_Float16*)alloc((size_t)128 * 64 * 2);
  _Float16* W2h   = (_Float16*)alloc((size_t)64 * 128 * 2);
  _Float16* xin   = (_Float16*)alloc(ROWS * IN_ * 2);
  _Float16* y1    = (_Float16*)alloc(ROWS * EMBH_ * 2);
  _Float16* xemb  = (_Float16*)alloc(ROWS * IN_ * 2);
  float*    gates = (float*)alloc((size_t)S_ * G4_ * 4);
  float*    cbuf  = (float*)alloc((size_t)S_ * HID_ * 4);
  _Float16* hbuf  = (_Float16*)alloc((size_t)S_ * HID_ * 2);
  _Float16* hcpad = (_Float16*)alloc((size_t)(S_ + 2) * HID_ * 2);
  _Float16* Hp    = (_Float16*)alloc(HR * HID_ * 2);
  _Float16* hb1   = (_Float16*)alloc(HR * 64 * 2);
  _Float16* hb2   = (_Float16*)alloc(HR * 128 * 2);
  _Float16* hb3   = (_Float16*)alloc(HR * 64 * 2);

  auto blk = [](size_t n) { return (unsigned)((n + 255) / 256); };
  auto gblk = [](size_t waves) { return (unsigned)((waves + 7) / 8); };

  // --- prologue: lane gate, packing, init -------------------------------
  k_lane<<<blk(S_), 256, 0, stream>>>(lane, lg_w1, lg_b1, lg_w2, lg_b2, lc);
  k_gbias<<<blk(G4_), 256, 0, stream>>>(b_ih, b_hh, gb);
  k_packwcat<<<blk((size_t)G4_ * KCAT_), 256, 0, stream>>>(W_ih, W_hh, Wcat);
  k_packconv<<<blk((size_t)3 * HID_ * HID_), 256, 0, stream>>>(conv_w, W3);
  k_pack16<<<blk(EMBH_ * IN_), 256, 0, stream>>>(emb_w1, We1, EMBH_ * IN_);
  k_pack16<<<blk(IN_ * EMBH_), 256, 0, stream>>>(emb_w2, We2, IN_ * EMBH_);
  k_pack16<<<blk(64 * HID_), 256, 0, stream>>>(out_w, Wo, 64 * HID_);
  k_pack16<<<blk(128 * 64), 256, 0, stream>>>(o1_w, W1h, 128 * 64);
  k_pack16<<<blk(64 * 128), 256, 0, stream>>>(o2_w, W2h, 64 * 128);
  k_scale<<<blk(ROWS * IN_), 256, 0, stream>>>(inputData, lc, xin);
  k_init<<<blk((size_t)S_ * HID_), 256, 0, stream>>>(hbuf, cbuf, hcpad);

  // --- embedding MLP (WMMA) ---------------------------------------------
  k_gemm<<<gblk((ROWS / 32) * (EMBH_ / 64)), 256, 0, stream>>>(
      xin, IN_, We1, IN_, emb_b1, y1, EMBH_, (int)ROWS, EMBH_, IN_, 1);
  k_gemm<<<gblk((ROWS / 32) * (IN_ / 64)), 256, 0, stream>>>(
      y1, EMBH_, We2, EMBH_, emb_b2, xemb, IN_, (int)ROWS, IN_, EMBH_, 1);

  // --- serial LSTM + spatial conv ---------------------------------------
  for (int t = 0; t < T_; ++t) {
    k_gates<<<gblk((S_ / 32) * (G4_ / 64)), 256, 0, stream>>>(
        xemb + (size_t)t * IN_, hbuf, Wcat, gb, gates);
    k_cell<<<blk((size_t)S_ * HID_), 256, 0, stream>>>(gates, cbuf, Cout, hcpad, t);
    k_conv<<<gblk((S_ / 32) * (HID_ / 64)), 256, 0, stream>>>(
        hcpad, W3, conv_b, hbuf, Hp, Hout, t);
  }

  // --- prediction head (WMMA) -------------------------------------------
  k_gemm<<<gblk((HR / 32) * (64 / 64)), 256, 0, stream>>>(
      Hp, HID_, Wo, HID_, out_b, hb1, 64, (int)HR, 64, HID_, 0);
  k_gemm<<<gblk((HR / 32) * (128 / 64)), 256, 0, stream>>>(
      hb1, 64, W1h, 64, o1_b, hb2, 128, (int)HR, 128, 64, 1);
  k_gemm<<<gblk((HR / 32) * (64 / 64)), 256, 0, stream>>>(
      hb2, 128, W2h, 128, o2_b, hb3, 64, (int)HR, 64, 128, 1);
  k_final<<<blk(HR), 256, 0, stream>>>(hb3, o3_w, o3_b, lc, out);
}
